// PointerGeneratorHead_31078383354026
// MI455X (gfx1250) — compile-verified, matrix operand order checked by reading, not run on hardware
//
#include <hip/hip_runtime.h>
#include <hip/hip_bf16.h>

#define NEG_INF_F (-1e9f)

// ---------------------------------------------------------------- dimensions
constexpr int Bb = 16, Ss = 512, Tt = 128, Hh = 1024, Ee = 512, Vv = 30000;
constexpr int ROWS  = Bb * Tt;   // 2048 decoded rows
constexpr int TROWS = Bb * Ss;   // 8192 text rows
constexpr int VNT = Vv / 16;     // 1875 vocab n-tiles
constexpr int VKT = Ee / 32;     // 16 k-steps for vocab GEMM

// ---------------------------------------------------------------- WMMA types
typedef __attribute__((ext_vector_type(8)))  float  v8f;
typedef __attribute__((ext_vector_type(16))) __bf16 v16bf;

union FragU {
  v16bf bf;
  uint4 q[2];
};

// Packed fragment layout (per 16x16(x32) tile): 32 lanes x 16 bf16 = 512 ushorts.
__device__ __forceinline__ v16bf load_frag(const unsigned short* __restrict__ tile,
                                           int lane) {
  FragU f;
  const uint4* p = reinterpret_cast<const uint4*>(tile + (lane << 4));
  f.q[0] = p[0];
  f.q[1] = p[1];
  return f.bf;
}

// Wave index as a compiler-provable scalar (keeps tile math in SGPRs and all
// tile-guard branches scalar, so EXEC stays all-ones around every WMMA).
__device__ __forceinline__ int wave_id() {
  return __builtin_amdgcn_readfirstlane((int)(threadIdx.x >> 5));
}

__device__ __forceinline__ v8f wmma_bf16(v16bf a, v16bf b, v8f c) {
  return __builtin_amdgcn_wmma_f32_16x16x32_bf16(false, a, false, b,
                                                 (short)0, c, false, false);
}

// ------------------------------------------------------- f32 -> bf16 packer
// mode 0: source is outer-major [Outer*16, K] (A fragments, or B fragments of
//         a transposed-source matrix -- identical bytes by layout symmetry).
// mode 1: source is k-major [K, Outer*16] (standard B fragments of [K,N]).
// Element e of lane l (row/col = l&15) maps to k = kt*32 + kmap(l,e):
//   kmap = (e<8 ? e : e+8) + ((l>>4)<<3)   (ISA 7.12.2, 16-bit A 16x32)
__global__ void pack_frags(const float* __restrict__ src,
                           unsigned short* __restrict__ dst,
                           int KT, int ldSrc, int mode,
                           long srcBatchStride, long elemsPerBatch, long total) {
  long id = (long)blockIdx.x * 256 + threadIdx.x;
  if (id >= total) return;
  long batch = id / elemsPerBatch;
  long rem   = id - batch * elemsPerBatch;
  int  e    = (int)(rem & 15);
  int  lane = (int)((rem >> 4) & 31);
  long tile = rem >> 9;
  int  kt    = (int)(tile % KT);
  long outer = tile / KT;
  int  ln = lane & 15;
  int  kk = (e < 8 ? e : e + 8) + ((lane >> 4) << 3);
  long k  = (long)kt * 32 + kk;
  const float* s = src + batch * srcBatchStride;
  float v = (mode == 0) ? s[(outer * 16 + ln) * (long)ldSrc + k]
                        : s[k * (long)ldSrc + outer * 16 + ln];
  unsigned u = __float_as_uint(v);
  unsigned r = (u + 0x7FFFu + ((u >> 16) & 1u)) >> 16;   // RNE f32->bf16
  dst[id] = (unsigned short)r;
}

// -------------------------------------------------- generic bf16 WMMA GEMM
// C[M,N] = scale * (A x B) + bias.  Block = 8 waves: 2 m-tiles x (4 waves x 4
// n-tiles) = 32x256 block tile.  Grids tile N EXACTLY, so the inner loop is
// guard-free; all 5 fragments per k-step are loaded before the WMMA burst so
// the scheduler can clause the loads and amortize one wait across 4 WMMAs.
__global__ void gemm_bf16(const unsigned short* __restrict__ Ap,
                          const unsigned short* __restrict__ Bp,
                          float* __restrict__ C, const float* __restrict__ bias,
                          int N, int KT, float scale,
                          long aBatchTiles, long bBatchTiles, long cBatchElems) {
  const int lane = threadIdx.x & 31;
  const int wave = wave_id();                      // scalar
  const int batch = blockIdx.z;
  const int mt  = blockIdx.y * 2 + (wave >> 2);    // scalar
  const int nt0 = blockIdx.x * 16 + (wave & 3) * 4;// scalar
  const unsigned short* A  = Ap + ((size_t)batch * aBatchTiles + (size_t)mt * KT) * 512;
  const unsigned short* Bm = Bp + ((size_t)batch * bBatchTiles + (size_t)nt0 * KT) * 512;
  v8f acc[4] = {};
#pragma unroll 2
  for (int kt = 0; kt < KT; ++kt) {
    v16bf a  = load_frag(A  + (size_t)kt * 512, lane);
    v16bf b0 = load_frag(Bm + ((size_t)0 * KT + kt) * 512, lane);
    v16bf b1 = load_frag(Bm + ((size_t)1 * KT + kt) * 512, lane);
    v16bf b2 = load_frag(Bm + ((size_t)2 * KT + kt) * 512, lane);
    v16bf b3 = load_frag(Bm + ((size_t)3 * KT + kt) * 512, lane);
    acc[0] = wmma_bf16(a, b0, acc[0]);
    acc[1] = wmma_bf16(a, b1, acc[1]);
    acc[2] = wmma_bf16(a, b2, acc[2]);
    acc[3] = wmma_bf16(a, b3, acc[3]);
  }
  float* Cb = C + (size_t)batch * cBatchElems;
  const int rbase = mt * 16 + ((lane >> 4) << 3);
#pragma unroll
  for (int j = 0; j < 4; ++j) {
    int col = (nt0 + j) * 16 + (lane & 15);
    float bv = bias ? bias[col] : 0.f;
#pragma unroll
    for (int r = 0; r < 8; ++r)
      Cb[(size_t)(rbase + r) * N + col] = acc[j][r] * scale + bv;
  }
}

// ---------------------------------------------- attention log-softmax over S
__global__ void attn_softmax(const float* __restrict__ score,
                             const unsigned char* __restrict__ tpad,
                             const unsigned char* __restrict__ dpad,
                             float* __restrict__ attnLog,
                             float* __restrict__ P) {
  const int row = blockIdx.x, tid = threadIdx.x;   // row = b*T + t
  const int b = row >> 7;                          // T = 128
  const float* sc = score + (size_t)row * Ss;
  float x0 = tpad[b * Ss + tid]       ? NEG_INF_F : sc[tid];
  float x1 = tpad[b * Ss + tid + 256] ? NEG_INF_F : sc[tid + 256];
  __shared__ float red[256];
  red[tid] = fmaxf(x0, x1); __syncthreads();
  for (int st = 128; st > 0; st >>= 1) {
    if (tid < st) red[tid] = fmaxf(red[tid], red[tid + st]);
    __syncthreads();
  }
  float M = red[0]; __syncthreads();
  red[tid] = __expf(x0 - M) + __expf(x1 - M); __syncthreads();
  for (int st = 128; st > 0; st >>= 1) {
    if (tid < st) red[tid] += red[tid + st];
    __syncthreads();
  }
  float l = M + logf(red[0]);
  bool dp = dpad[row] != 0;
  float a0 = dp ? NEG_INF_F : x0 - l;
  float a1 = dp ? NEG_INF_F : x1 - l;
  attnLog[(size_t)row * Ss + tid]       = a0;
  attnLog[(size_t)row * Ss + tid + 256] = a1;
  P[(size_t)row * Ss + tid]       = __expf(a0);
  P[(size_t)row * Ss + tid + 256] = __expf(a1);
}

// --------------------------------------- pointer-gate: before / log-sigmoid
__global__ void gate_kernel(const float* __restrict__ attended,
                            const float* __restrict__ dec,
                            const float* __restrict__ emb,
                            const float* __restrict__ Wp,
                            const float* __restrict__ bp,
                            float* __restrict__ genprob,
                            float* __restrict__ copyprob) {
  const int row = blockIdx.x, tid = threadIdx.x;
  const float* ar = attended + (size_t)row * Hh;
  const float* dr = dec      + (size_t)row * Hh;
  const float* er = emb      + (size_t)row * Ee;
  float sum = 0.f;
  for (int i = tid; i < Hh; i += 256) sum += ar[i] * Wp[i] + dr[i] * Wp[Hh + i];
  for (int i = tid; i < Ee; i += 256) sum += er[i] * Wp[2 * Hh + i];
  __shared__ float red[256];
  red[tid] = sum; __syncthreads();
  for (int st = 128; st > 0; st >>= 1) {
    if (tid < st) red[tid] += red[tid + st];
    __syncthreads();
  }
  if (tid == 0) {
    float x  = red[0] + bp[0];
    float gp = fminf(x, 0.f) - log1pf(__expf(-fabsf(x)));  // log_sigmoid
    genprob[row]  = gp;
    copyprob[row] = gp - x;                                // log(1-sigmoid)
  }
}

// ------------------- vocab GEMM pass 1: online (max, sumexp), logits stay in
// registers.  The 16-frag A panel for this 16-row block is preloaded into
// VGPRs once (128 VGPRs) -- the n-loop then only streams B from L2.
__global__ void vocab_stats(const unsigned short* __restrict__ Ap,
                            const unsigned short* __restrict__ Bp,
                            const float* __restrict__ bg,
                            float2* __restrict__ partials,
                            int nchunks, int tilesPerChunk) {
  const int tid = threadIdx.x;
  const int lane = tid & 31;
  const int wave = wave_id();                      // scalar
  const int mt = blockIdx.y, chunk = blockIdx.x;
  const int t0 = chunk * tilesPerChunk;
  int t1 = t0 + tilesPerChunk; if (t1 > VNT) t1 = VNT;
  const unsigned short* A = Ap + (size_t)mt * VKT * 512;
  v16bf afr[VKT];
#pragma unroll
  for (int kt = 0; kt < VKT; ++kt) afr[kt] = load_frag(A + (size_t)kt * 512, lane);
  float m[8], s[8];
#pragma unroll
  for (int r = 0; r < 8; ++r) { m[r] = -1e30f; s[r] = 0.f; }
  for (int nt = t0 + wave; nt < t1; nt += 8) {     // scalar loop bounds
    const unsigned short* Btile = Bp + (size_t)nt * VKT * 512;
    if (nt + 8 < t1)
      __builtin_prefetch(Bp + (size_t)(nt + 8) * VKT * 512, 0, 1);
    v8f acc = {};
#pragma unroll
    for (int kt = 0; kt < VKT; ++kt) {
      v16bf b = load_frag(Btile + (size_t)kt * 512, lane);
      acc = wmma_bf16(afr[kt], b, acc);
    }
    float bv = bg[nt * 16 + (lane & 15)];
#pragma unroll
    for (int r = 0; r < 8; ++r) {
      float c  = acc[r] + bv;
      float mn = fmaxf(m[r], c);
      s[r] = s[r] * __expf(m[r] - mn) + __expf(c - mn);
      m[r] = mn;
    }
  }
  __shared__ float2 red[256][8];
#pragma unroll
  for (int r = 0; r < 8; ++r) red[tid][r] = make_float2(m[r], s[r]);
  __syncthreads();
  if (tid < 16) {            // one thread per row of this 16-row block
    const int j = tid & 7, lg = tid >> 3;
    float M = -1e30f, S = 0.f;
    for (int wv = 0; wv < 8; ++wv)
      for (int l = 0; l < 16; ++l) {
        float2 p  = red[wv * 32 + lg * 16 + l][j];
        float  mn = fmaxf(M, p.x);
        S = S * __expf(M - mn) + p.y * __expf(p.x - mn);
        M = mn;
      }
    partials[((size_t)mt * nchunks + chunk) * 16 + tid] = make_float2(M, S);
  }
}

__global__ void vocab_reduce(const float2* __restrict__ partials, int nchunks,
                             float* __restrict__ lse) {
  int r = blockIdx.x * 256 + threadIdx.x;
  if (r >= ROWS) return;
  int mt = r >> 4, wi = r & 15;
  float M = -1e30f, S = 0.f;
  for (int c = 0; c < nchunks; ++c) {
    float2 p  = partials[((size_t)mt * nchunks + c) * 16 + wi];
    float  mn = fmaxf(M, p.x);
    S = S * __expf(M - mn) + p.y * __expf(p.x - mn);
    M = mn;
  }
  lse[r] = M + logf(S);
}

// -------- vocab GEMM pass 2: recompute tile (Wg lives in L2), write
// gen = exp(logit - lse + gen_prob) directly into d_out.
__global__ void vocab_write(const unsigned short* __restrict__ Ap,
                            const unsigned short* __restrict__ Bp,
                            const float* __restrict__ bg,
                            const float* __restrict__ lse,
                            const float* __restrict__ genprob,
                            float* __restrict__ out) {
  const int lane = threadIdx.x & 31;
  const int wave = wave_id();                      // scalar
  const int mt = blockIdx.y;
  const int nt = blockIdx.x * 8 + wave;            // scalar guard below
  if (nt >= VNT) return;
  const unsigned short* A     = Ap + (size_t)mt * VKT * 512;
  const unsigned short* Btile = Bp + (size_t)nt * VKT * 512;
  v8f acc = {};
#pragma unroll
  for (int kt = 0; kt < VKT; ++kt) {
    v16bf a = load_frag(A + (size_t)kt * 512, lane);
    v16bf b = load_frag(Btile + (size_t)kt * 512, lane);
    acc = wmma_bf16(a, b, acc);
  }
  const int col   = nt * 16 + (lane & 15);
  const float bv  = bg[col];
  const int rbase = mt * 16 + ((lane >> 4) << 3);
#pragma unroll
  for (int r = 0; r < 8; ++r) {
    int row = rbase + r;
    out[(size_t)row * Vv + col] = __expf(acc[r] + bv - lse[row] + genprob[row]);
  }
}

// --------------------------- copy distribution scatter-add into vocab space
__global__ void scatter_copy(const float* __restrict__ attnLog,
                             const float* __restrict__ copyprob,
                             const int* __restrict__ label,
                             float* __restrict__ out) {
  long id = (long)blockIdx.x * 256 + threadIdx.x;   // over B*T*S
  int  s  = (int)(id & (Ss - 1));
  long bt = id >> 9;
  int  b  = (int)(bt >> 7);
  float v = attnLog[id] + copyprob[bt];
  if (v > -80.f)
    atomicAdd(out + bt * (long)Vv + label[b * Ss + s], __expf(v));
}

// ------------------------------------------------------- final log + mask
__global__ void finalize_log(float* __restrict__ out) {
  size_t id = (size_t)blockIdx.x * 256 + threadIdx.x;
  float v = out[id];
  float l = (v > 0.f) ? logf(v) : NEG_INF_F;
  if (!(l >= -1e9f && l <= 1e9f)) l = NEG_INF_F;
  out[id] = l;
}

// ============================================================== host driver
extern "C" void kernel_launch(void* const* d_in, const int* in_sizes, int n_in,
                              void* d_out, int out_size, void* d_ws, size_t ws_size,
                              hipStream_t stream) {
  const float*         text  = (const float*)d_in[0];
  const unsigned char* tpad  = (const unsigned char*)d_in[1];
  const float*         dec   = (const float*)d_in[2];
  const unsigned char* dpad  = (const unsigned char*)d_in[3];
  const float*         emb   = (const float*)d_in[4];
  const int*           label = (const int*)d_in[5];
  const float* Wq = (const float*)d_in[6];   const float* bq = (const float*)d_in[7];
  const float* Wk = (const float*)d_in[8];   const float* bk = (const float*)d_in[9];
  const float* Wh = (const float*)d_in[10];  const float* bh = (const float*)d_in[11];
  const float* Wg = (const float*)d_in[12];  const float* bg = (const float*)d_in[13];
  const float* Wp = (const float*)d_in[14];  const float* bp = (const float*)d_in[15];
  float* out = (float*)d_out;

  char* w = (char*)d_ws;
  size_t off = 0;
  auto alloc = [&](size_t bytes) -> void* {
    void* p = w + off;
    off += (bytes + 255) & ~(size_t)255;
    return p;
  };
  unsigned short* decA   = (unsigned short*)alloc((size_t)ROWS * Hh * 2);
  unsigned short* textA  = (unsigned short*)alloc((size_t)TROWS * Hh * 2);
  unsigned short* WqB    = (unsigned short*)alloc((size_t)Hh * Hh * 2);
  unsigned short* WkB    = (unsigned short*)alloc((size_t)Hh * Hh * 2);
  unsigned short* WhB    = (unsigned short*)alloc((size_t)Hh * Ee * 2);
  unsigned short* WgB    = (unsigned short*)alloc((size_t)Ee * Vv * 2);
  float*          qbuf   = (float*)alloc((size_t)ROWS * Hh * 4);
  float*          kbuf   = (float*)alloc((size_t)TROWS * Hh * 4);
  float*          dembuf = (float*)alloc((size_t)ROWS * Ee * 4);
  unsigned short* qA     = (unsigned short*)alloc((size_t)ROWS * Hh * 2);
  unsigned short* kA     = (unsigned short*)alloc((size_t)TROWS * Hh * 2);
  unsigned short* dembA  = (unsigned short*)alloc((size_t)ROWS * Ee * 2);
  float*          score  = (float*)alloc((size_t)ROWS * Ss * 4);
  float*          attnL  = (float*)alloc((size_t)ROWS * Ss * 4);
  float*          Pbuf   = (float*)alloc((size_t)ROWS * Ss * 4);
  unsigned short* PA     = (unsigned short*)alloc((size_t)ROWS * Ss * 2);
  unsigned short* textB  = (unsigned short*)alloc((size_t)TROWS * Hh * 2);
  float*          attd   = (float*)alloc((size_t)ROWS * Hh * 4);
  float*          genp   = (float*)alloc((size_t)ROWS * 4);
  float*          copyp  = (float*)alloc((size_t)ROWS * 4);
  constexpr int NCHUNK = 30, TPC = 63;   // 30*63 = 1890 >= 1875 tiles
  float2*         parts  = (float2*)alloc((size_t)(ROWS / 16) * NCHUNK * 16 * 8);
  float*          lse    = (float*)alloc((size_t)ROWS * 4);

  auto cdiv = [](long a, long b) { return (a + b - 1) / b; };
  const float* nobias = nullptr;

  // ---- stage 0: pack everything into WMMA fragment layout (f32 -> bf16)
  { long t = (long)ROWS * Hh;
    pack_frags<<<cdiv(t,256),256,0,stream>>>(dec,  decA,  32, Hh, 0, 0, t, t); }
  { long t = (long)TROWS * Hh;
    pack_frags<<<cdiv(t,256),256,0,stream>>>(text, textA, 32, Hh, 0, 0, t, t); }
  { long t = (long)Hh * Hh;
    pack_frags<<<cdiv(t,256),256,0,stream>>>(Wq, WqB, 32, Hh, 1, 0, t, t);
    pack_frags<<<cdiv(t,256),256,0,stream>>>(Wk, WkB, 32, Hh, 1, 0, t, t); }
  { long t = (long)Hh * Ee;
    pack_frags<<<cdiv(t,256),256,0,stream>>>(Wh, WhB, 32, Ee, 1, 0, t, t); }
  { long t = (long)Ee * Vv;
    pack_frags<<<cdiv(t,256),256,0,stream>>>(Wg, WgB, 16, Vv, 1, 0, t, t); }

  // ---- stage 1: projections (q, k, dec_emb)
  gemm_bf16<<<dim3(4,  64, 1), 256, 0, stream>>>(decA,  WqB, qbuf,   bq, Hh, 32, 1.f, 0, 0, 0);
  gemm_bf16<<<dim3(4, 256, 1), 256, 0, stream>>>(textA, WkB, kbuf,   bk, Hh, 32, 1.f, 0, 0, 0);
  gemm_bf16<<<dim3(2,  64, 1), 256, 0, stream>>>(decA,  WhB, dembuf, bh, Ee, 32, 1.f, 0, 0, 0);

  // repack activations as fragments (kA doubles as B operand of q @ k^T)
  { long t = (long)ROWS * Hh;
    pack_frags<<<cdiv(t,256),256,0,stream>>>(qbuf, qA, 32, Hh, 0, 0, t, t); }
  { long t = (long)TROWS * Hh;
    pack_frags<<<cdiv(t,256),256,0,stream>>>(kbuf, kA, 32, Hh, 0, 0, t, t); }
  { long t = (long)ROWS * Ee;
    pack_frags<<<cdiv(t,256),256,0,stream>>>(dembuf, dembA, 16, Ee, 0, 0, t, t); }

  // ---- stage 2: batched attention score = q @ k^T / 32
  gemm_bf16<<<dim3(2, 4, Bb), 256, 0, stream>>>(qA, kA, score, nobias,
      Ss, 32, 0.03125f, /*aTiles*/ 8 * 32, /*bTiles*/ 32 * 32, (long)Tt * Ss);

  attn_softmax<<<ROWS, 256, 0, stream>>>(score, tpad, dpad, attnL, Pbuf);

  // ---- stage 3: attended = exp(attn) @ text
  { long t = (long)ROWS * Ss;
    pack_frags<<<cdiv(t,256),256,0,stream>>>(Pbuf, PA, 16, Ss, 0, 0, t, t); }
  { long t = (long)TROWS * Hh;   // per-batch B fragments of text[b] (K=S, N=H)
    pack_frags<<<cdiv(t,256),256,0,stream>>>(text, textB, 16, Hh, 1,
        (long)Ss * Hh, (long)Ss * Hh, t); }
  gemm_bf16<<<dim3(4, 4, Bb), 256, 0, stream>>>(PA, textB, attd, nobias,
      Hh, 16, 1.f, /*aTiles*/ 8 * 16, /*bTiles*/ 64 * 16, (long)Tt * Hh);

  // ---- stage 4: pointer gate
  gate_kernel<<<ROWS, 256, 0, stream>>>(attd, dec, emb, Wp, bp, genp, copyp);

  // ---- stage 5: vocab head, two-pass online log-softmax (no logits tensor)
  vocab_stats<<<dim3(NCHUNK, ROWS / 16), 256, 0, stream>>>(dembA, WgB, bg,
                                                           parts, NCHUNK, TPC);
  vocab_reduce<<<ROWS / 256, 256, 0, stream>>>(parts, NCHUNK, lse);
  vocab_write<<<dim3((VNT + 7) / 8, ROWS / 16), 256, 0, stream>>>(
      dembA, WgB, bg, lse, genp, out);

  // ---- stage 6: scatter copy distribution, then log + finite mask
  scatter_copy<<<(ROWS * Ss) / 256, 256, 0, stream>>>(attnL, copyp, label, out);
  finalize_log<<<(long)ROWS * Vv / 256, 256, 0, stream>>>(out);
}
